// AttnResModule2D_58162447123346
// MI455X (gfx1250) — matrix operand order repaired and checked by reading, not compiled, not run on hardware
//
#include <hip/hip_runtime.h>

// Problem constants (match reference harness)
#define L_HIST 12
#define NL     13      // L+1
#define NB     4
#define NS     1024
#define ND     1024
#define LN_EPS 1e-5f

typedef __attribute__((ext_vector_type(16))) __bf16        v16bf;
typedef __attribute__((ext_vector_type(8)))  float          v8f;
typedef __attribute__((ext_vector_type(8)))  unsigned short us8;

static __device__ __forceinline__ unsigned short f32_to_bf16(float f) {
  unsigned int u = __builtin_bit_cast(unsigned int, f);
  u += 0x7FFFu + ((u >> 16) & 1u);           // round to nearest even
  return (unsigned short)(u >> 16);
}

// ---------------------------------------------------------------------------
// Kernel 0: global scalars  swg = sum(w*gamma),  swb = sum(w*beta)
// ---------------------------------------------------------------------------
__global__ __launch_bounds__(256) void wdot_kernel(const float* __restrict__ w,
                                                   const float* __restrict__ g,
                                                   const float* __restrict__ be,
                                                   float* __restrict__ outs) {
  __shared__ float r0[256], r1[256];
  float a = 0.f, c = 0.f;
  for (int i = threadIdx.x; i < ND; i += 256) { a += w[i] * g[i]; c += w[i] * be[i]; }
  r0[threadIdx.x] = a; r1[threadIdx.x] = c; __syncthreads();
  for (int off = 128; off; off >>= 1) {
    if (threadIdx.x < (unsigned)off) {
      r0[threadIdx.x] += r0[threadIdx.x + off];
      r1[threadIdx.x] += r1[threadIdx.x + off];
    }
    __syncthreads();
  }
  if (threadIdx.x == 0) { outs[0] = r0[0]; outs[1] = r1[0]; }
}

// ---------------------------------------------------------------------------
// Kernel 1: fused LayerNorm + score.  One wave per row of length D=1024.
// score = rstd*(dot(w*gamma, v) - mu*swg) + swb   (LN collapsed algebraically)
// ---------------------------------------------------------------------------
__global__ __launch_bounds__(256) void ln_scores_kernel(const float* __restrict__ hist,
                                                        const float* __restrict__ cur,
                                                        const float* __restrict__ w,
                                                        const float* __restrict__ g,
                                                        const float* __restrict__ swgb,
                                                        float* __restrict__ scores) {
  const int wave = threadIdx.x >> 5, lane = threadIdx.x & 31;
  const int r   = blockIdx.x * 8 + wave;   // 0 .. 13*4*1024-1, ordered [l][b][s]
  const int l   = r >> 12;                 // / (B*S)
  const int rem = r & 4095;
  const int b   = rem >> 10;
  const int s   = rem & 1023;
  const float* src = (l < L_HIST) ? (hist + (size_t)r * ND)
                                  : (cur  + (size_t)rem * ND);
  float sv = 0.f, sv2 = 0.f, swv = 0.f;
#pragma unroll
  for (int c = 0; c < 8; ++c) {
    int d = c * 128 + lane * 4;
    float4 x  = *(const float4*)(src + d);
    float4 wv = *(const float4*)(w + d);
    float4 gv = *(const float4*)(g + d);
    sv  += x.x + x.y + x.z + x.w;
    sv2 += x.x * x.x + x.y * x.y + x.z * x.z + x.w * x.w;
    swv += wv.x * gv.x * x.x + wv.y * gv.y * x.y + wv.z * gv.z * x.z + wv.w * gv.w * x.w;
  }
#pragma unroll
  for (int off = 16; off; off >>= 1) {     // wave32 reduction
    sv  += __shfl_xor(sv,  off, 32);
    sv2 += __shfl_xor(sv2, off, 32);
    swv += __shfl_xor(swv, off, 32);
  }
  if (lane == 0) {
    float swg = swgb[0], swb = swgb[1];
    float mu   = sv * (1.0f / ND);
    float var  = sv2 * (1.0f / ND) - mu * mu;
    float rstd = rsqrtf(var + LN_EPS);
    scores[((b * NL + l) << 10) + s] = rstd * (swv - mu * swg) + swb;   // [B][13][S]
  }
}

// ---------------------------------------------------------------------------
// Kernel 2: per (b,t) flash stats over masked set {scores[b,l,s] : s<=t}
// ---------------------------------------------------------------------------
__global__ __launch_bounds__(256) void softmax_stats_kernel(const float* __restrict__ scores,
                                                            float* __restrict__ Mout,
                                                            float* __restrict__ Ziout) {
  const int t = blockIdx.x, b = blockIdx.y;
  const float* sc = scores + (size_t)b * NL * NS;
  __shared__ float red[256];
  float m = -3.402823466e38f;
  for (int l = 0; l < NL; ++l) {
    const float* row = sc + l * NS;
    for (int s = threadIdx.x; s <= t; s += 256) m = fmaxf(m, row[s]);
  }
  red[threadIdx.x] = m; __syncthreads();
  for (int off = 128; off; off >>= 1) {
    if (threadIdx.x < (unsigned)off)
      red[threadIdx.x] = fmaxf(red[threadIdx.x], red[threadIdx.x + off]);
    __syncthreads();
  }
  m = red[0]; __syncthreads();
  float z = 0.f;
  for (int l = 0; l < NL; ++l) {
    const float* row = sc + l * NS;
    for (int s = threadIdx.x; s <= t; s += 256) z += __expf(row[s] - m);
  }
  red[threadIdx.x] = z; __syncthreads();
  for (int off = 128; off; off >>= 1) {
    if (threadIdx.x < (unsigned)off) red[threadIdx.x] += red[threadIdx.x + off];
    __syncthreads();
  }
  if (threadIdx.x == 0) { Mout[b * NS + t] = m; Ziout[b * NS + t] = 1.0f / red[0]; }
}

// ---------------------------------------------------------------------------
// Kernel 3: h[b,t,d] = scale[t] * sum_{l,s<=t} exp(score[b,l,s]-Mblk) * V[l,b,s,d]
// where Mblk = M[T0+127] (block-wide max; M is monotone in t) and
// scale[t] = exp(Mblk - M[t] + log(1/Z[t])).
// Fully-causal K-chunks have an A-matrix with IDENTICAL rows -> A fragment
// comes straight from a 32-entry LDS vector (no per-row exp, no P tile).
// Only the 4 diagonal chunks/layer take the masked per-row path.
// Block tile: 128 t x 128 d, 8 waves (4x2), wave tile 32x64
// = 2x4 v_wmma_f32_16x16x32_bf16 accumulators. K chunk = 32.
// ---------------------------------------------------------------------------
__global__ __launch_bounds__(256) void pv_wmma_kernel(const float* __restrict__ hist,
                                                      const float* __restrict__ cur,
                                                      const float* __restrict__ scores,
                                                      const float* __restrict__ Mbuf,
                                                      const float* __restrict__ Zibuf,
                                                      float* __restrict__ out) {
  const int T0 = blockIdx.x * 128;
  const int D0 = blockIdx.y * 128;
  const int b  = blockIdx.z;
  const int tid  = threadIdx.x;
  const int lane = tid & 31, wave = tid >> 5;
  const int wmrow = wave & 3;    // 4 wave-rows x 32 t  = 128
  const int wncol = wave >> 2;   // 2 wave-cols x 64 d  = 128

  __shared__ __align__(32) unsigned short sP[128 * 32];  // masked P tile (diag chunks)
  __shared__ __align__(32) unsigned short sV[128 * 32];  // V tile bf16, [d][s] (K-major)
  __shared__ __align__(32) unsigned short sE[32];        // shared A row e_k (causal chunks)
  __shared__ float sScale[128];                          // per-row epilogue scale

  const float Mblk = Mbuf[b * NS + T0 + 127];            // >= every score in range
  if (tid < 128) {
    float mt  = Mbuf[b * NS + T0 + tid];
    float zit = Zibuf[b * NS + T0 + tid];
    sScale[tid] = __expf((Mblk - mt) + __logf(zit));     // log-space: no overflow
  }
  __syncthreads();

  v8f acc[2][4] = {};

  const int kk    = tid & 31;      // P-stage: key column
  const int rgrp  = tid >> 5;      // P-stage: 16-row group
  const int s_end = (T0 + 128 < NS) ? (T0 + 128) : NS;   // causal chunk skip
  const int m16   = lane & 15;
  const int khalf = lane >> 4;

  struct AB { us8 lo, hi; };

  for (int l = 0; l < NL; ++l) {
    const float* vbase = (l < L_HIST) ? (hist + ((size_t)l * NB + b) * NS * ND)
                                      : (cur  + (size_t)b * NS * ND);
    const float* srow  = scores + ((size_t)b * NL + l) * NS;

    // ---------------- phase 1: fully causal chunks (s0+31 < T0) ------------
    for (int s0 = 0; s0 < T0; s0 += 32) {
      __syncthreads();
      if (tid < 32) sE[tid] = f32_to_bf16(__expf(srow[s0 + tid] - Mblk));
      if (s0 + 32 < s_end)   // cover next chunk's latency (global_prefetch_b8)
        __builtin_prefetch(&vbase[(size_t)(s0 + 32 + (tid >> 3)) * ND + D0 + (tid & 7) * 16], 0, 1);
#pragma unroll
      for (int i = 0; i < 16; ++i) {          // stage V [128 d x 32 s] -> bf16
        int idx = tid + i * 256;
        int ss  = idx >> 7, dd = idx & 127;
        sV[dd * 32 + ss] = f32_to_bf16(vbase[(size_t)(s0 + ss) * ND + (D0 + dd)]);
      }
      __syncthreads();

      AB u;                                   // A rows identical: read from sE
      u.lo = *(const us8*)(const void*)&sE[khalf * 16];
      u.hi = *(const us8*)(const void*)&sE[khalf * 16 + 8];
      v16bf afrag = __builtin_bit_cast(v16bf, u);
#pragma unroll
      for (int mi = 0; mi < 2; ++mi)
#pragma unroll
        for (int nj = 0; nj < 4; ++nj) {
          int col = wncol * 64 + nj * 16 + m16;
          v16bf bfrag = *(const v16bf*)(const void*)&sV[col * 32 + khalf * 16];
          acc[mi][nj] = __builtin_amdgcn_wmma_f32_16x16x32_bf16(
              false, afrag, false, bfrag, (short)0, acc[mi][nj], false, false);
        }
    }

    // ---------------- phase 2: diagonal (masked) chunks --------------------
    for (int s0 = T0; s0 < s_end; s0 += 32) {
      __syncthreads();
      {
        float e = __expf(srow[s0 + kk] - Mblk);
        unsigned short eb = f32_to_bf16(e);
#pragma unroll
        for (int i = 0; i < 16; ++i) {        // per-row causal select only
          int tl = rgrp * 16 + i;
          sP[tl * 32 + kk] = ((s0 + kk) <= (T0 + tl)) ? eb : (unsigned short)0;
        }
      }
#pragma unroll
      for (int i = 0; i < 16; ++i) {
        int idx = tid + i * 256;
        int ss  = idx >> 7, dd = idx & 127;
        sV[dd * 32 + ss] = f32_to_bf16(vbase[(size_t)(s0 + ss) * ND + (D0 + dd)]);
      }
      __syncthreads();

#pragma unroll
      for (int mi = 0; mi < 2; ++mi) {
        int row = wmrow * 32 + mi * 16 + m16;
        AB u;
        u.lo = *(const us8*)(const void*)&sP[row * 32 + khalf * 16];
        u.hi = *(const us8*)(const void*)&sP[row * 32 + khalf * 16 + 8];
        v16bf afrag = __builtin_bit_cast(v16bf, u);
#pragma unroll
        for (int nj = 0; nj < 4; ++nj) {
          int col = wncol * 64 + nj * 16 + m16;
          v16bf bfrag = *(const v16bf*)(const void*)&sV[col * 32 + khalf * 16];
          acc[mi][nj] = __builtin_amdgcn_wmma_f32_16x16x32_bf16(
              false, afrag, false, bfrag, (short)0, acc[mi][nj], false, false);
        }
      }
    }
  }

  // ---- epilogue: per-row rescale (softmax renorm) and store ----
#pragma unroll
  for (int mi = 0; mi < 2; ++mi)
#pragma unroll
    for (int nj = 0; nj < 4; ++nj) {
      int col = D0 + wncol * 64 + nj * 16 + (lane & 15);
#pragma unroll
      for (int r = 0; r < 8; ++r) {
        int tl = wmrow * 32 + mi * 16 + r + ((lane >> 4) << 3);  // C layout M=r(+8)
        out[((size_t)b * NS + (T0 + tl)) * (size_t)ND + col] = acc[mi][nj][r] * sScale[tl];
      }
    }
}

// ---------------------------------------------------------------------------
extern "C" void kernel_launch(void* const* d_in, const int* in_sizes, int n_in,
                              void* d_out, int out_size, void* d_ws, size_t ws_size,
                              hipStream_t stream) {
  const float* hist = (const float*)d_in[0];   // [12,4,1024,1024]
  const float* cur  = (const float*)d_in[1];   // [4,1024,1024]
  const float* w    = (const float*)d_in[2];   // [1024]
  const float* g    = (const float*)d_in[3];   // gamma [1024]
  const float* be   = (const float*)d_in[4];   // beta  [1024]
  float* out = (float*)d_out;                  // [4,1024,1024]

  float* f      = (float*)d_ws;
  float* swgb   = f;                           // 2 scalars (padded to 16)
  float* scores = f + 16;                      // 13*4*1024
  float* Mbuf   = scores + NL * NB * NS;       // 4*1024
  float* Zibuf  = Mbuf + NB * NS;              // 4*1024

  wdot_kernel<<<1, 256, 0, stream>>>(w, g, be, swgb);
  ln_scores_kernel<<<(NL * NB * NS) / 8, 256, 0, stream>>>(hist, cur, w, g, swgb, scores);
  softmax_stats_kernel<<<dim3(NS, NB), 256, 0, stream>>>(scores, Mbuf, Zibuf);
  pv_wmma_kernel<<<dim3(NS / 128, ND / 128, NB), 256, 0, stream>>>(hist, cur, scores,
                                                                   Mbuf, Zibuf, out);
}
